// GraphSELayer_49237505081486
// MI455X (gfx1250) — matrix operand order repaired and checked by reading, not compiled, not run on hardware
//
#include <hip/hip_runtime.h>
#include <hip/hip_bf16.h>

typedef _Float16 f16;
typedef __attribute__((ext_vector_type(16))) _Float16 v16h;
typedef __attribute__((ext_vector_type(8)))  float    v8f;
typedef __attribute__((ext_vector_type(4)))  float    v4f;

#define GG 1024   // num graphs
#define CC 256    // channels
#define RR 16     // bottleneck dim (C/R)

// ---------------------------------------------------------------------------
// Kernel 1: segment mean over sorted batch_idx. One block per graph,
// thread c owns channel c. Deterministic (no atomics): binary search the
// contiguous row range for this graph, then coalesced accumulate.
// x is streamed once -> non-temporal loads keep batch_idx/score cache-warm.
// ---------------------------------------------------------------------------
__global__ void seg_mean_kernel(const float* __restrict__ x,
                                const int* __restrict__ bidx,
                                int n, f16* __restrict__ mean) {
  const int g = blockIdx.x;
  const int c = threadIdx.x;
  __shared__ int s_lo, s_hi;
  if (c == 0) {
    int lo = 0, hi = n;
    while (lo < hi) { int mid = (lo + hi) >> 1; if (bidx[mid] <  g    ) lo = mid + 1; else hi = mid; }
    s_lo = lo;
    lo = 0; hi = n;
    while (lo < hi) { int mid = (lo + hi) >> 1; if (bidx[mid] <  g + 1) lo = mid + 1; else hi = mid; }
    s_hi = lo;
  }
  __syncthreads();
  const int lo = s_lo, hi = s_hi;
  float acc = 0.0f;
  for (int i = lo; i < hi; ++i)
    acc += __builtin_nontemporal_load(&x[(size_t)i * CC + c]);
  const float cnt = (float)(hi - lo);
  const float m = acc / fmaxf(cnt, 1.0f);
  mean[g * CC + c] = (f16)m;
}

// ---------------------------------------------------------------------------
// Kernel 2: H = Xmean @ W1^T with fused PReLU.  [1024,256] x [256,16] -> [1024,16]
// One wave (32 lanes) per 16-graph tile; 8 x v_wmma_f32_16x16x32_f16 over K=256.
// Fragment layouts per CDNA5 ISA 7.12.2 (16-bit A 16x32, B 32x16, f32 C/D 16x16).
// ---------------------------------------------------------------------------
__global__ void gemm1_kernel(const f16* __restrict__ mean,
                             const float* __restrict__ W1,
                             const float* __restrict__ prelu_a,
                             f16* __restrict__ h) {
  const int gt   = blockIdx.x;        // graph tile, 0..63
  const int lane = threadIdx.x;       // 0..31
  const int m    = lane & 15;         // A row (and B column index)
  const int hi   = lane >> 4;         // lane-half select
  const float ap = prelu_a[0];

  v8f acc = {};
  #pragma unroll
  for (int kt = 0; kt < CC; kt += 32) {
    v16h a, b;
    #pragma unroll
    for (int j = 0; j < 8; ++j) {
      // A 16x32: lanes 0-15 row m hold K {0..7,16..23}; lanes 16-31 {8..15,24..31}
      const int ka = ((j < 4) ? 0 : 16) + hi * 8 + 2 * (j & 3);
      a[2*j]   = mean[(gt * 16 + m) * CC + kt + ka];
      a[2*j+1] = mean[(gt * 16 + m) * CC + kt + ka + 1];
      // B 32x16: same K order as A (lanes 0-15 K {0..7,16..23}; lanes 16-31 {8..15,24..31}), col n=lane&15
      const int kb = ((j < 4) ? 0 : 16) + hi * 8 + 2 * (j & 3);
      b[2*j]   = (f16)W1[m * CC + kt + kb];       // B[k][n] = W1[n][k], n == m
      b[2*j+1] = (f16)W1[m * CC + kt + kb + 1];
    }
    acc = __builtin_amdgcn_wmma_f32_16x16x32_f16(false, a, false, b,
                                                 (short)0, acc, false, false);
  }
  // D 16x16 f32: VGPR v -> row (v + 8*hi), col (lane&15). Fused PReLU, store f16.
  #pragma unroll
  for (int v = 0; v < 8; ++v) {
    const int row = gt * 16 + v + 8 * hi;
    float val = acc[v];
    val = (val >= 0.0f) ? val : ap * val;
    h[row * RR + m] = (f16)val;
  }
}

// ---------------------------------------------------------------------------
// Kernel 3: score = sigmoid(H' @ W2^T). [1024,16] x [16,256] -> [1024,256]
// K=16 real, zero-padded to 32. One wave per 16-graph tile, 16 column tiles.
// ---------------------------------------------------------------------------
__global__ void gemm2_kernel(const f16* __restrict__ h,
                             const float* __restrict__ W2,
                             float* __restrict__ score) {
  const int gt   = blockIdx.x;
  const int lane = threadIdx.x;
  const int m    = lane & 15;
  const int hi   = lane >> 4;

  // A fragment: slots j<4 carry K = hi*8 + {0..7} (all real, K<16); j>=4 -> K>=16 -> 0
  v16h a = {};
  #pragma unroll
  for (int j = 0; j < 4; ++j) {
    const int k0 = hi * 8 + 2 * j;
    a[2*j]   = h[(gt * 16 + m) * RR + k0];
    a[2*j+1] = h[(gt * 16 + m) * RR + k0 + 1];
  }

  for (int ct = 0; ct < CC / 16; ++ct) {
    // B fragment: same K order as A; slots j<4 carry K = hi*8 + 2(j&3) (real), j>=4 -> K>=16 -> 0
    v16h b = {};
    #pragma unroll
    for (int j = 0; j < 8; ++j) {
      const int kb = ((j < 4) ? 0 : 16) + hi * 8 + 2 * (j & 3);
      const bool ok = (kb + 1) < RR;   // true iff j<4
      b[2*j]   = ok ? (f16)W2[(ct * 16 + m) * RR + kb]     : (f16)0.0f;
      b[2*j+1] = ok ? (f16)W2[(ct * 16 + m) * RR + kb + 1] : (f16)0.0f;
    }
    v8f acc = {};
    acc = __builtin_amdgcn_wmma_f32_16x16x32_f16(false, a, false, b,
                                                 (short)0, acc, false, false);
    #pragma unroll
    for (int v = 0; v < 8; ++v) {
      const int row = gt * 16 + v + 8 * hi;
      const float s = 1.0f / (1.0f + __expf(-acc[v]));
      score[row * CC + ct * 16 + m] = s;
    }
  }
}

// ---------------------------------------------------------------------------
// Kernel 4: out = x * score[batch_idx], float4-vectorized.
// x/out are 512MB streams -> non-temporal; score (1MB) + batch_idx (2MB) stay in L2.
// ---------------------------------------------------------------------------
__global__ void apply_kernel(const float* __restrict__ x,
                             const int* __restrict__ bidx,
                             const float* __restrict__ score,
                             float* __restrict__ out, long long nquad) {
  const long long t = (long long)blockIdx.x * blockDim.x + threadIdx.x;
  if (t >= nquad) return;
  const int i  = (int)(t >> 6);   // CC/4 = 64 quads per row
  const int c4 = (int)(t & 63);
  const int g  = bidx[i];
  const v4f xv = __builtin_nontemporal_load((const v4f*)x + t);
  const v4f sv = ((const v4f*)score)[(size_t)g * 64 + c4];
  __builtin_nontemporal_store(xv * sv, (v4f*)out + t);
}

// ---------------------------------------------------------------------------
extern "C" void kernel_launch(void* const* d_in, const int* in_sizes, int n_in,
                              void* d_out, int out_size, void* d_ws, size_t ws_size,
                              hipStream_t stream) {
  const float* x    = (const float*)d_in[0];
  const int*   bidx = (const int*)  d_in[1];
  // d_in[2] = num_graphs (fixed at GG=1024)
  const float* W1   = (const float*)d_in[3];
  const float* pa   = (const float*)d_in[4];
  const float* W2   = (const float*)d_in[5];
  float* out = (float*)d_out;
  const int n = in_sizes[0] / CC;   // number of nodes

  // Workspace layout: score f32 [G*C] | mean f16 [G*C] | h f16 [G*R]
  char* ws = (char*)d_ws;
  float* score = (float*)ws;
  f16*   mean  = (f16*)(ws + (size_t)GG * CC * sizeof(float));
  f16*   hbuf  = (f16*)(ws + (size_t)GG * CC * sizeof(float)
                           + (size_t)GG * CC * sizeof(f16));

  seg_mean_kernel<<<GG, CC, 0, stream>>>(x, bidx, n, mean);
  gemm1_kernel<<<GG / 16, 32, 0, stream>>>(mean, W1, pa, hbuf);
  gemm2_kernel<<<GG / 16, 32, 0, stream>>>(hbuf, W2, score);

  const long long nquad = (long long)n * (CC / 4);
  const int threads = 256;
  const unsigned blocks = (unsigned)((nquad + threads - 1) / threads);
  apply_kernel<<<blocks, threads, 0, stream>>>(x, bidx, score, out, nquad);
}
